// PlugableIPAdapter_11269994184731
// MI455X (gfx1250) — matrix operand order repaired and checked by reading, not compiled, not run on hardware
//
#include <hip/hip_runtime.h>
#include <hip/hip_bf16.h>

// ---------------------------------------------------------------------------
// MI455X (gfx1250) IP-Adapter cross-attention, bf16 WMMA pipeline.
//   1) Q = x @ Wq                    (WMMA GEMM, fp32 in -> bf16 out)
//   2) K/V = context @ Wk/Wv,  Kip/Vip = image_emb @ Wk_ip/Wv_ip
//   3) fused dual attention (kv=77 + kv=16), softmax in registers,
//      logits never touch HBM; K/V/Q tiles via async global->LDS copies
//   4) out = attn @ Wo + bo          (WMMA GEMM, bf16 A, fp32 out + bias)
// GEMM: 128x256 block tile, 64x64 wave tile (16 WMMA / 8 frag loads),
//       B tile stored transposed in LDS (contiguous fragment loads),
//       ping-pong LDS double buffering (1 barrier / K-step).
// ---------------------------------------------------------------------------

typedef __bf16 bf16;
typedef bf16  v16bf __attribute__((ext_vector_type(16)));
typedef float v8f   __attribute__((ext_vector_type(8)));
typedef int   v4i   __attribute__((ext_vector_type(4)));

#define AS1 __attribute__((address_space(1)))
#define AS3 __attribute__((address_space(3)))

#if __has_builtin(__builtin_amdgcn_global_load_async_to_lds_b128) && \
    __has_builtin(__builtin_amdgcn_s_wait_asynccnt)
#define ASYNC_LDS 1
#else
#define ASYNC_LDS 0
#endif

__device__ inline void cp_async16(void* lds_dst, const void* gsrc) {
#if ASYNC_LDS
  __builtin_amdgcn_global_load_async_to_lds_b128((AS1 v4i*)gsrc,
                                                 (AS3 v4i*)lds_dst, 0, 0);
#else
  *(int4*)lds_dst = *(const int4*)gsrc;
#endif
}

__device__ inline void cp_async_wait() {
#if ASYNC_LDS
  __builtin_amdgcn_s_wait_asynccnt(0);
#endif
}

__device__ inline v8f vzero8() {
  v8f z;
#pragma unroll
  for (int i = 0; i < 8; ++i) z[i] = 0.0f;
  return z;
}

__device__ inline v8f wmma_bf16(v16bf a, v16bf b, v8f c) {
  // D = A(16x32 bf16) * B(32x16 bf16) + C(16x16 f32)
  return __builtin_amdgcn_wmma_f32_16x16x32_bf16(
      /*neg_a=*/false, a, /*neg_b=*/false, b,
      /*c_mod=*/(short)0, c, /*reuse_a=*/false, /*reuse_b=*/false);
}

// A fragment: rows mrow..mrow+15, contraction cols 0..31 of `base` (row stride ld).
// Lanes 0-15: M=lane, elems 0-15 = K0-15.
// Lanes 16-31: M=lane-16, elems 0-15 = K16-31.
__device__ inline v16bf load_a_frag(const bf16* base, int ld, int mrow, int lane) {
  int m  = mrow + (lane & 15);
  int kb = (lane < 16) ? 0 : 16;
  const bf16* p = base + m * ld + kb;
  v16bf a;
#pragma unroll
  for (int i = 0; i < 16; ++i) a[i] = p[i];
  return a;
}

// B fragment from tile stored [n][contraction] (row stride ld) — i.e. the
// contraction dim is contiguous. Lanes 0-15: N=lane, K0-15; lanes 16-31:
// N=lane-16, K16-31. Vectorizes to 2x ds_load_b128 per lane.
__device__ inline v16bf load_bT_frag(const bf16* base, int ld, int ncol, int lane) {
  int n  = ncol + (lane & 15);
  int kb = (lane < 16) ? 0 : 16;
  const bf16* p = base + n * ld + kb;
  v16bf b;
#pragma unroll
  for (int i = 0; i < 16; ++i) b[i] = p[i];
  return b;
}

// ---------------------------------------------------------------------------
// Tiled WMMA GEMM: C[M,N] = A[M,K] * B[K,N] (+bias).
// 256 thr = 8 waves (2x4 wave grid), block tile 128x256, K-step 32,
// wave tile 64x64 (4x4 fragments). B tile stored TRANSPOSED in LDS.
// Ping-pong LDS double buffering: one barrier per K-step.
// ---------------------------------------------------------------------------
template <bool A_BF16, bool OUT_BF16, bool ADD_BIAS>
__global__ __launch_bounds__(256) void gemm_wmma_kernel(
    const void* __restrict__ Aptr, const float* __restrict__ Bw,
    const float* __restrict__ bias, void* __restrict__ Cptr,
    int M, int N, int K) {
  constexpr int BM = 128, BN = 256, BK = 32;
  constexpr int LDA  = BK + 8;  // 40 bf16 elems
  constexpr int LDBT = BK + 8;  // 40 bf16 elems
  __shared__ bf16 Al[2][BM * LDA];    // 2 x 10240 B
  __shared__ bf16 Blt[2][BN * LDBT];  // 2 x 20480 B  (total 60 KB)

  const int tid  = threadIdx.x;
  const int lane = tid & 31;
  const int wave = tid >> 5;
  const int wm   = wave >> 2;  // 0..1
  const int wn   = wave & 3;   // 0..3
  const int m0   = blockIdx.y * BM;
  const int n0   = blockIdx.x * BN;

  const float* Af = (const float*)Aptr;
  const bf16*  Ab = (const bf16*)Aptr;

  auto load_tile = [&](int k0, int st) {
    {  // A tile: 128x32 row-major, each thread one 16-elem row segment
      int row = tid >> 1;
      int cs  = (tid & 1) * 16;
      int gm  = m0 + row;
      bf16* dst = &Al[st][row * LDA + cs];
      if (gm < M) {
        if constexpr (A_BF16) {
          const bf16* src = Ab + (size_t)gm * K + k0 + cs;
#pragma unroll
          for (int i = 0; i < 16; ++i) dst[i] = src[i];
        } else {
          const float* src = Af + (size_t)gm * K + k0 + cs;
#pragma unroll
          for (int i = 0; i < 16; ++i) dst[i] = (bf16)src[i];
        }
      } else {
#pragma unroll
        for (int i = 0; i < 16; ++i) dst[i] = (bf16)0.0f;
      }
    }
    {  // B tile, stored transposed: lane owns column n = tid, walks k.
       // Global reads coalesced across lanes; LDS writes merge to b64.
      const float* src = Bw + (size_t)k0 * N + n0 + tid;
      bf16* dst = &Blt[st][tid * LDBT];
#pragma unroll
      for (int kk = 0; kk < BK; ++kk) dst[kk] = (bf16)src[(size_t)kk * N];
    }
  };

  v8f acc[4][4];
#pragma unroll
  for (int i = 0; i < 4; ++i)
#pragma unroll
    for (int j = 0; j < 4; ++j) acc[i][j] = vzero8();

  const int nk = K / BK;
  load_tile(0, 0);
  __syncthreads();

  for (int kt = 0; kt < nk; ++kt) {
    const int st = kt & 1;
    if (kt + 1 < nk) load_tile((kt + 1) * BK, st ^ 1);  // prefetch next stage

    v16bf af[4], bfr[4];
#pragma unroll
    for (int i = 0; i < 4; ++i)
      af[i] = load_a_frag(&Al[st][0], LDA, wm * 64 + i * 16, lane);
#pragma unroll
    for (int j = 0; j < 4; ++j)
      bfr[j] = load_bT_frag(&Blt[st][0], LDBT, wn * 64 + j * 16, lane);
#pragma unroll
    for (int i = 0; i < 4; ++i)
#pragma unroll
      for (int j = 0; j < 4; ++j) acc[i][j] = wmma_bf16(af[i], bfr[j], acc[i][j]);
    __syncthreads();
  }

  // Store: lane holds col (lane&15), rows (lane>=16 ? 8:0)+r per C-layout.
  const int nc = lane & 15;
  const int mh = (lane >> 4) * 8;
#pragma unroll
  for (int i = 0; i < 4; ++i) {
#pragma unroll
    for (int j = 0; j < 4; ++j) {
      int col = n0 + wn * 64 + j * 16 + nc;
      float bv = 0.0f;
      if constexpr (ADD_BIAS) bv = bias[col];
#pragma unroll
      for (int r = 0; r < 8; ++r) {
        int row = m0 + wm * 64 + i * 16 + mh + r;
        if (row < M) {
          float v = acc[i][j][r] + bv;
          if constexpr (OUT_BF16)
            ((bf16*)Cptr)[(size_t)row * N + col] = (bf16)v;
          else
            ((float*)Cptr)[(size_t)row * N + col] = v;
        }
      }
    }
  }
}

// ---------------------------------------------------------------------------
// Fused dual cross-attention. One block = one (b, h, 128-query tile).
// kv_main = 77 (padded to 80, masked), kv_ip = 16; concatenated to 96 so a
// single P@V pass yields  softmax(QK^T/8)V + softmax(QKip^T/8)Vip.
// All tiles arrive via async global->LDS b128 copies (ASYNCcnt).
// ---------------------------------------------------------------------------
__global__ __launch_bounds__(256) void ip_attn_kernel(
    const bf16* __restrict__ Qb, const bf16* __restrict__ Kb,
    const bf16* __restrict__ Vb, const bf16* __restrict__ Kip,
    const bf16* __restrict__ Vip, bf16* __restrict__ Outb) {
  constexpr int DM = 1280, DH = 64, SKV = 77, SIP = 16, QT = 128;
  constexpr int LQ = 72;   // row stride (bf16): 144 B = 9 x 16 B (b128-aligned)
  constexpr int LP = 104;  // prob tile row stride: 208 B = 13 x 16 B

  // Aliased LDS (barriers separate phases):
  //  [0      ,13824) Vc   [96][72]   (main V rows 0-79, ip V rows 80-95)
  //  [13824  ,32256) Qs   [128][72]  -- phase 1 only
  //  [32256  ,43776) Ks   [80][72]   -- phase 1 only
  //  [43776  ,46080) Kips [16][72]   -- phase 1 only
  //  [13824  ,40448) Pc   [128][104] -- phase 2, reuses Qs/Ks region
  __shared__ __align__(16) char smem[46080];
  bf16* Vc   = (bf16*)smem;
  bf16* Qs   = (bf16*)(smem + 13824);
  bf16* Ks   = (bf16*)(smem + 32256);
  bf16* Kips = (bf16*)(smem + 43776);
  bf16* Pc   = (bf16*)(smem + 13824);

  const int tid = threadIdx.x, lane = tid & 31, wave = tid >> 5;
  const int b = blockIdx.z, h = blockIdx.y;
  const int q0 = blockIdx.x * QT;
  const size_t qrow0 = (size_t)b * 4096 + q0;
  const int hc = h * DH;

  // ---- async tile loads: pure bf16 copies, 16 B per op ----
  for (int idx = tid; idx < QT * 8; idx += 256) {    // Q tile (raw; scale later)
    int m = idx >> 3, c8 = (idx & 7) * 8;
    cp_async16(&Qs[m * LQ + c8], &Qb[(qrow0 + m) * DM + hc + c8]);
  }
  for (int idx = tid; idx < SKV * 8; idx += 256) {   // K + V main
    int k = idx >> 3, c8 = (idx & 7) * 8;
    cp_async16(&Ks[k * LQ + c8], &Kb[((size_t)b * SKV + k) * DM + hc + c8]);
    cp_async16(&Vc[k * LQ + c8], &Vb[((size_t)b * SKV + k) * DM + hc + c8]);
  }
  for (int idx = tid; idx < SIP * 8; idx += 256) {   // Kip + Vip
    int k = idx >> 3, c8 = (idx & 7) * 8;
    cp_async16(&Kips[k * LQ + c8], &Kip[((size_t)b * SIP + k) * DM + hc + c8]);
    cp_async16(&Vc[(80 + k) * LQ + c8], &Vip[((size_t)b * SIP + k) * DM + hc + c8]);
  }
  for (int idx = tid; idx < 3 * DH; idx += 256) {    // zero pad keys 77..79
    int k = 77 + idx / DH, c = idx % DH;
    Ks[k * LQ + c] = (bf16)0.0f;
    Vc[k * LQ + c] = (bf16)0.0f;
  }
  cp_async_wait();
  __syncthreads();

  // ---- logits: each wave owns 16 query rows, all 96 key columns ----
  v8f sm[5], sip;
#pragma unroll
  for (int f = 0; f < 5; ++f) sm[f] = vzero8();
  sip = vzero8();

#pragma unroll
  for (int ks = 0; ks < 64; ks += 32) {
    v16bf aq = load_a_frag(Qs + ks, LQ, wave * 16, lane);
#pragma unroll
    for (int f = 0; f < 5; ++f) {
      v16bf bk = load_bT_frag(Ks + ks, LQ, f * 16, lane);
      sm[f] = wmma_bf16(aq, bk, sm[f]);
    }
    v16bf bi = load_bT_frag(Kips + ks, LQ, 0, lane);
    sip = wmma_bf16(aq, bi, sip);
  }
  // scale 1/sqrt(64), then mask padded keys 77..79 (frag 4, nc in {13,14,15})
#pragma unroll
  for (int f = 0; f < 5; ++f)
#pragma unroll
    for (int r = 0; r < 8; ++r) sm[f][r] *= 0.125f;
#pragma unroll
  for (int r = 0; r < 8; ++r) sip[r] *= 0.125f;
  if ((lane & 15) >= 13) {
#pragma unroll
    for (int r = 0; r < 8; ++r) sm[4][r] = -1e30f;
  }
  __syncthreads();  // done reading Qs/Ks/Kips; Pc may now overwrite them

  // ---- softmax in registers (rows spread over one 16-lane half) ----
  const int prow = wave * 16 + ((lane >> 4) ? 8 : 0);
#pragma unroll
  for (int r = 0; r < 8; ++r) {
    // main (77 keys)
    float mx = sm[0][r];
#pragma unroll
    for (int f = 1; f < 5; ++f) mx = fmaxf(mx, sm[f][r]);
    for (int off = 8; off; off >>= 1) mx = fmaxf(mx, __shfl_xor(mx, off, 32));
    float e[5], sum = 0.0f;
#pragma unroll
    for (int f = 0; f < 5; ++f) { e[f] = __expf(sm[f][r] - mx); sum += e[f]; }
    for (int off = 8; off; off >>= 1) sum += __shfl_xor(sum, off, 32);
    float inv = 1.0f / sum;
    int row = prow + r;
#pragma unroll
    for (int f = 0; f < 5; ++f)
      Pc[row * LP + f * 16 + (lane & 15)] = (bf16)(e[f] * inv);
    // ip (16 keys), IP_WEIGHT = 1.0 folded into concatenation
    float mi = sip[r];
    for (int off = 8; off; off >>= 1) mi = fmaxf(mi, __shfl_xor(mi, off, 32));
    float ei = __expf(sip[r] - mi);
    float si = ei;
    for (int off = 8; off; off >>= 1) si += __shfl_xor(si, off, 32);
    Pc[row * LP + 80 + (lane & 15)] = (bf16)(ei / si);
  }
  __syncthreads();

  // ---- out = Pc[128x96] @ Vc[96x64]  (main + ip fused) ----
  v8f o[4];
#pragma unroll
  for (int f = 0; f < 4; ++f) o[f] = vzero8();
#pragma unroll
  for (int ks = 0; ks < 96; ks += 32) {
    v16bf ap = load_a_frag(Pc + ks, LP, wave * 16, lane);
#pragma unroll
    for (int f = 0; f < 4; ++f) {
      // V stored [key][d]: contraction is the key index -> strided gather,
      // but only 12 fragments total per wave here (cheap vs the GEMMs).
      int n = f * 16 + (lane & 15);
      int kb = (lane < 16) ? 0 : 16;
      v16bf bv;
#pragma unroll
      for (int i = 0; i < 16; ++i) bv[i] = Vc[(ks + kb + i) * LQ + n];
      o[f] = wmma_bf16(ap, bv, o[f]);
    }
  }

  // ---- store merged-head bf16 [B*S, 1280] ----
  const int nc = lane & 15;
  const int mh = (lane >> 4) * 8;
#pragma unroll
  for (int f = 0; f < 4; ++f) {
    int col = hc + f * 16 + nc;
#pragma unroll
    for (int r = 0; r < 8; ++r) {
      int m = wave * 16 + mh + r;
      Outb[(qrow0 + m) * DM + col] = (bf16)o[f][r];
    }
  }
}

// ---------------------------------------------------------------------------
extern "C" void kernel_launch(void* const* d_in, const int* in_sizes, int n_in,
                              void* d_out, int out_size, void* d_ws, size_t ws_size,
                              hipStream_t stream) {
  (void)in_sizes; (void)n_in; (void)out_size; (void)ws_size;
  const float* x         = (const float*)d_in[0];
  const float* context   = (const float*)d_in[1];
  const float* image_emb = (const float*)d_in[2];
  const float* Wq        = (const float*)d_in[3];
  const float* Wk        = (const float*)d_in[4];
  const float* Wv        = (const float*)d_in[5];
  const float* Wkip      = (const float*)d_in[6];
  const float* Wvip      = (const float*)d_in[7];
  const float* Wo        = (const float*)d_in[8];
  const float* bo        = (const float*)d_in[9];
  float* out = (float*)d_out;

  const int B = 4, S = 4096, Dm = 1280, C = 2048, T = 16, SKV = 77;
  const int MQ = B * S;  // 16384

  char* ws = (char*)d_ws;
  size_t off = 0;
  auto alloc = [&](size_t bytes) {
    char* p = ws + off;
    off += (bytes + 255) & ~(size_t)255;
    return p;
  };
  bf16* Qb   = (bf16*)alloc((size_t)MQ * Dm * 2);       // 40 MB
  bf16* Kbuf = (bf16*)alloc((size_t)B * SKV * Dm * 2);  // 0.8 MB
  bf16* Vbuf = (bf16*)alloc((size_t)B * SKV * Dm * 2);
  bf16* Kipb = (bf16*)alloc((size_t)B * T * Dm * 2);
  bf16* Vipb = (bf16*)alloc((size_t)B * T * Dm * 2);
  bf16* Attn = (bf16*)alloc((size_t)MQ * Dm * 2);       // 40 MB

  dim3 blk(256);
  dim3 gq(Dm / 256, MQ / 128);                 // 5 x 128
  dim3 gkv(Dm / 256, (B * SKV + 127) / 128);   // 5 x 3
  dim3 gip(Dm / 256, 1);                       // 5 x 1
  dim3 ga(S / 128, 20, B);                     // 32 x 20 x 4

  // 1) Q projection
  gemm_wmma_kernel<false, true, false><<<gq, blk, 0, stream>>>(x, Wq, nullptr, Qb, MQ, Dm, Dm);
  // 2) context K/V projections
  gemm_wmma_kernel<false, true, false><<<gkv, blk, 0, stream>>>(context, Wk, nullptr, Kbuf, B * SKV, Dm, C);
  gemm_wmma_kernel<false, true, false><<<gkv, blk, 0, stream>>>(context, Wv, nullptr, Vbuf, B * SKV, Dm, C);
  // 3) image-embedding K/V projections
  gemm_wmma_kernel<false, true, false><<<gip, blk, 0, stream>>>(image_emb, Wkip, nullptr, Kipb, B * T, Dm, C);
  gemm_wmma_kernel<false, true, false><<<gip, blk, 0, stream>>>(image_emb, Wvip, nullptr, Vipb, B * T, Dm, C);
  // 4) fused dual attention
  ip_attn_kernel<<<ga, blk, 0, stream>>>(Qb, Kbuf, Vbuf, Kipb, Vipb, Attn);
  // 5) output projection + bias -> fp32 d_out
  gemm_wmma_kernel<true, false, true><<<gq, blk, 0, stream>>>(Attn, Wo, bo, out, MQ, Dm, Dm);
}
